// LinearAttention3D_47047071760494
// MI455X (gfx1250) — matrix operand order repaired and checked by reading, not compiled, not run on hardware
//
#include <hip/hip_runtime.h>
#include <math.h>

typedef unsigned short ushort_t;
typedef __attribute__((ext_vector_type(16))) __bf16 v16bf;
typedef __attribute__((ext_vector_type(8)))  float  v8f;
typedef __attribute__((ext_vector_type(4)))  unsigned int u32x4;

#define Bb 2
#define Cc 128
#define Nn 131072            // 32*64*64
#define HEADSc 4
#define DHEAD 32
#define HIDc 128
#define MQKV 384
#define GROUPSc 32
#define EPSc 1e-5f
#define NSLICES 128
#define NB 128               // N columns per GEMM workgroup
#define LSTR 136             // padded LDS row stride (ushorts): 272B, 16B aligned

union FragBF { v16bf bf; u32x4 q[2]; };

__device__ __forceinline__ ushort_t f2bf(float f) {
  union { float f; unsigned u; } v; v.f = f;
  unsigned u = v.u;
  u += 0x7FFFu + ((u >> 16) & 1u);           // round to nearest even
  return (ushort_t)(u >> 16);
}
__device__ __forceinline__ float bf2f(ushort_t h) {
  union { unsigned u; float f; } v; v.u = ((unsigned)h) << 16;
  return v.f;
}

// ---------------- GroupNorm statistics: one block per (b, group) ----------------
__global__ void gn_stats_kernel(const float* __restrict__ X, float* __restrict__ stats) {
  int blk = blockIdx.x;
  int b = blk >> 5, g = blk & 31;
  const float* base = X + ((long long)b * Cc + g * 4) * Nn;  // 4 contiguous channels
  const int total = 4 * Nn;
  float s = 0.f, ss = 0.f;
  for (int i = threadIdx.x; i < total; i += 256) {
    float v = base[i];
    s += v; ss += v * v;
  }
  __shared__ float rs[256], rss[256];
  rs[threadIdx.x] = s; rss[threadIdx.x] = ss;
  __syncthreads();
  for (int off = 128; off > 0; off >>= 1) {
    if (threadIdx.x < off) { rs[threadIdx.x] += rs[threadIdx.x + off]; rss[threadIdx.x] += rss[threadIdx.x + off]; }
    __syncthreads();
  }
  if (threadIdx.x == 0) {
    float mean = rs[0] / (float)total;
    float var  = rss[0] / (float)total - mean * mean;
    stats[blk * 2]     = mean;
    stats[blk * 2 + 1] = rsqrtf(var + EPSc);
  }
}

// -------- Fold GroupNorm (per-batch) into qkv weights: w' bf16, bias' f32 --------
__global__ void prep_qkv_w_kernel(const float* __restrict__ qkv_w, const float* __restrict__ norm_w,
                                  const float* __restrict__ norm_b, const float* __restrict__ stats,
                                  ushort_t* __restrict__ wbf, float* __restrict__ bias) {
  int idx = blockIdx.x * blockDim.x + threadIdx.x;     // b*384 + o
  if (idx >= Bb * MQKV) return;
  int b = idx / MQKV, o = idx % MQKV;
  const float* wr = qkv_w + (long long)o * Cc;
  ushort_t* wo = wbf + (long long)idx * Cc;
  float bacc = 0.f;
  for (int c = 0; c < Cc; ++c) {
    int g = c >> 2;
    float mean = stats[(b * 32 + g) * 2];
    float rstd = stats[(b * 32 + g) * 2 + 1];
    float sc = rstd * norm_w[c];
    float t  = norm_b[c] - mean * sc;
    wo[c] = f2bf(wr[c] * sc);
    bacc += wr[c] * t;
  }
  bias[idx] = bacc;
}

// ---------------- bf16 WMMA GEMM:  Out[b][M][N] = W[b][M][128] x X[b][128][N] + bias ----------------
__global__ void __launch_bounds__(256)
wmma_gemm_kernel(const ushort_t* __restrict__ Wbf,   // [B][M][128] bf16 (K contiguous)
                 const float* __restrict__ bias,     // [B][M]
                 const void* __restrict__ Xsrc, int x_is_f32, long long x_bstride,
                 void* __restrict__ Out, int out_is_f32, long long o_bstride,
                 int M) {
  __shared__ ushort_t xt[NB * LSTR];      // transposed input tile [n][c]
  __shared__ ushort_t wt[64 * LSTR];      // weight chunk [m][c]
  int b = blockIdx.y;
  long long n0 = (long long)blockIdx.x * NB;
  int tid = threadIdx.x;

  if (x_is_f32) {
    const float* X = (const float*)Xsrc + (long long)b * x_bstride;
    for (int i = tid; i < NB * Cc; i += 256) {
      int c = i >> 7, n = i & 127;
      xt[n * LSTR + c] = f2bf(X[(long long)c * Nn + n0 + n]);
    }
  } else {
    const ushort_t* X = (const ushort_t*)Xsrc + (long long)b * x_bstride;
    for (int i = tid; i < NB * Cc; i += 256) {
      int c = i >> 7, n = i & 127;
      xt[n * LSTR + c] = X[(long long)c * Nn + n0 + n];
    }
  }
  __syncthreads();

  int wave = tid >> 5, lane = tid & 31;
  int hf = lane >> 4, l15 = lane & 15;
  int ncol = (wave << 4) + l15;                      // this wave's output column in block

  // B fragments (K = 128 -> 4 chunks of 32), held across all M chunks.
  // B 32x16 bf16 layout: lane = column, elements are K = kc*32 + 16*hf + i (contiguous 16)
  FragBF bfr[4];
  #pragma unroll
  for (int kc = 0; kc < 4; ++kc) {
    const ushort_t* p = &xt[ncol * LSTR + kc * 32 + 16 * hf];
    bfr[kc].q[0] = *(const u32x4*)p;
    bfr[kc].q[1] = *(const u32x4*)(p + 8);
  }

  for (int m0 = 0; m0 < M; m0 += 64) {
    __syncthreads();
    const ushort_t* Wg = Wbf + ((long long)b * M + m0) * Cc;
    for (int i = tid; i < 64 * Cc; i += 256) {
      int m = i >> 7, c = i & 127;
      wt[m * LSTR + c] = Wg[i];
    }
    __syncthreads();
    #pragma unroll
    for (int mt = 0; mt < 4; ++mt) {
      v8f acc;
      const float* bp = bias + (long long)b * M + m0 + mt * 16 + 8 * hf;
      #pragma unroll
      for (int r = 0; r < 8; ++r) acc[r] = bp[r];
      #pragma unroll
      for (int kc = 0; kc < 4; ++kc) {
        // A 16x32 bf16 layout: lane = row, contiguous 16-element K run at 16*hf
        FragBF af;
        const ushort_t* p = &wt[(mt * 16 + l15) * LSTR + kc * 32 + 16 * hf];
        af.q[0] = *(const u32x4*)p;
        af.q[1] = *(const u32x4*)(p + 8);
        acc = __builtin_amdgcn_wmma_f32_16x16x32_bf16(false, af.bf, false, bfr[kc].bf,
                                                      (short)0, acc, false, false);
      }
      long long obase = (long long)(m0 + mt * 16 + 8 * hf) * Nn + n0 + ncol;
      if (out_is_f32) {
        float* O = (float*)Out + (long long)b * o_bstride + obase;
        #pragma unroll
        for (int r = 0; r < 8; ++r) O[(long long)r * Nn] = acc[r];
      } else {
        ushort_t* O = (ushort_t*)Out + (long long)b * o_bstride + obase;
        #pragma unroll
        for (int r = 0; r < 8; ++r) O[(long long)r * Nn] = f2bf(acc[r]);
      }
    }
  }
}

// ---------------- k softmax over d (32 strided elems), in place on bf16 ----------------
__global__ void ksoftmax_kernel(ushort_t* __restrict__ qkv) {
  long long id = (long long)blockIdx.x * blockDim.x + threadIdx.x;  // (b*4+h)*N + n
  long long n = id & (Nn - 1);
  int bh = (int)(id >> 17);
  int b = bh >> 2, h = bh & 3;
  ushort_t* base = qkv + ((long long)b * MQKV + HIDc + h * DHEAD) * Nn + n;
  float v[32];
  float mx = -1e30f;
  #pragma unroll
  for (int d = 0; d < 32; ++d) { v[d] = bf2f(base[(long long)d * Nn]); mx = fmaxf(mx, v[d]); }
  float s = 0.f;
  #pragma unroll
  for (int d = 0; d < 32; ++d) { v[d] = __expf(v[d] - mx); s += v[d]; }
  float inv = 1.f / s;
  #pragma unroll
  for (int d = 0; d < 32; ++d) base[(long long)d * Nn] = f2bf(v[d] * inv);
}

// ---------------- q softmax row stats (max, sumexp) over N per row ----------------
__global__ void qstats_kernel(const ushort_t* __restrict__ qkv, float* __restrict__ qstats) {
  int row = blockIdx.x;                 // b*128 + hd
  int b = row >> 7, hd = row & 127;
  const ushort_t* base = qkv + ((long long)b * MQKV + hd) * Nn;
  __shared__ float red[256];
  float mx = -1e30f;
  for (int i = threadIdx.x; i < Nn; i += 256) mx = fmaxf(mx, bf2f(base[i]));
  red[threadIdx.x] = mx; __syncthreads();
  for (int off = 128; off > 0; off >>= 1) {
    if (threadIdx.x < off) red[threadIdx.x] = fmaxf(red[threadIdx.x], red[threadIdx.x + off]);
    __syncthreads();
  }
  mx = red[0];
  __syncthreads();
  float s = 0.f;
  for (int i = threadIdx.x; i < Nn; i += 256) s += __expf(bf2f(base[i]) - mx);
  red[threadIdx.x] = s; __syncthreads();
  for (int off = 128; off > 0; off >>= 1) {
    if (threadIdx.x < off) red[threadIdx.x] += red[threadIdx.x + off];
    __syncthreads();
  }
  if (threadIdx.x == 0) { qstats[row * 2] = mx; qstats[row * 2 + 1] = red[0]; }
}

// ---------------- q normalize in place ----------------
__global__ void qnorm_kernel(ushort_t* __restrict__ qkv, const float* __restrict__ qstats) {
  long long id = (long long)blockIdx.x * blockDim.x + threadIdx.x;  // b*2^24 + hd*2^17 + n
  int b = (int)(id >> 24);
  long long rem = id & ((1 << 24) - 1);
  int hd = (int)(rem >> 17);
  long long n = rem & (Nn - 1);
  float mx  = qstats[(b * 128 + hd) * 2];
  float inv = 1.f / qstats[(b * 128 + hd) * 2 + 1];
  ushort_t* p = qkv + ((long long)b * MQKV + hd) * Nn + n;
  *p = f2bf(__expf(bf2f(*p) - mx) * inv);
}

// ---------------- ctx partials: ctx[d,e] = sum_n k'[d,n]*v[e,n], per N-slice ----------------
__global__ void __launch_bounds__(256)
ctx_partial_kernel(const ushort_t* __restrict__ qkv, float* __restrict__ part) {
  int slice = blockIdx.x;               // 0..NSLICES-1
  int bh = blockIdx.y;                  // 0..7
  int b = bh >> 2, h = bh & 3;
  int tid = threadIdx.x, wave = tid >> 5, lane = tid & 31;
  int hf = lane >> 4, l15 = lane & 15;
  const long long colsPerSlice = Nn / NSLICES;       // 1024
  const long long colsPerWave  = colsPerSlice / 8;   // 128
  long long nbase0 = (long long)slice * colsPerSlice + (long long)wave * colsPerWave;
  const ushort_t* kbase = qkv + ((long long)b * MQKV + HIDc + h * DHEAD) * Nn;
  const ushort_t* vbase = qkv + ((long long)b * MQKV + 2 * HIDc + h * DHEAD) * Nn;

  v8f acc[2][2];
  #pragma unroll
  for (int i = 0; i < 2; ++i)
    #pragma unroll
    for (int j = 0; j < 2; ++j)
      #pragma unroll
      for (int r = 0; r < 8; ++r) acc[i][j][r] = 0.f;

  for (long long nb = nbase0; nb < nbase0 + colsPerWave; nb += 32) {
    FragBF ka[2], vb[2];
    #pragma unroll
    for (int t = 0; t < 2; ++t) {
      // A frag: rows d = t*16 + l15 of k'; contiguous K run at nb + 16*hf
      const ushort_t* pa = kbase + (long long)(t * 16 + l15) * Nn + nb + 16 * hf;
      ka[t].q[0] = *(const u32x4*)pa;
      ka[t].q[1] = *(const u32x4*)(pa + 8);
      // B frag: cols e = t*16 + l15 of v; contiguous K run at nb + 16*hf
      const ushort_t* pb = vbase + (long long)(t * 16 + l15) * Nn + nb + 16 * hf;
      vb[t].q[0] = *(const u32x4*)pb;
      vb[t].q[1] = *(const u32x4*)(pb + 8);
    }
    #pragma unroll
    for (int mt = 0; mt < 2; ++mt)
      #pragma unroll
      for (int nt = 0; nt < 2; ++nt)
        acc[mt][nt] = __builtin_amdgcn_wmma_f32_16x16x32_bf16(false, ka[mt].bf, false, vb[nt].bf,
                                                              (short)0, acc[mt][nt], false, false);
  }

  __shared__ float ctxp[8][32][32];
  #pragma unroll
  for (int mt = 0; mt < 2; ++mt)
    #pragma unroll
    for (int nt = 0; nt < 2; ++nt)
      #pragma unroll
      for (int r = 0; r < 8; ++r)
        ctxp[wave][mt * 16 + 8 * hf + r][nt * 16 + l15] = acc[mt][nt][r];
  __syncthreads();
  for (int j = tid; j < 1024; j += 256) {
    float s = 0.f;
    #pragma unroll
    for (int w = 0; w < 8; ++w) s += ctxp[w][j >> 5][j & 31];
    part[((long long)bh * NSLICES + slice) * 1024 + j] = s;
  }
}

// ---------------- deterministic slice reduction ----------------
__global__ void ctx_reduce_kernel(const float* __restrict__ part, float* __restrict__ ctx) {
  int j = blockIdx.x * blockDim.x + threadIdx.x;    // 0..8*1024-1
  int bh = j >> 10, e = j & 1023;
  float s = 0.f;
  for (int sl = 0; sl < NSLICES; ++sl) s += part[((long long)bh * NSLICES + sl) * 1024 + e];
  ctx[j] = s;
}

// -------- Fold ctx into out projection: W2[b][c][h*32+d] = sum_e out_w[c][h*32+e]*ctx[b,h][d][e] --------
__global__ void prep_out_w_kernel(const float* __restrict__ out_w, const float* __restrict__ out_b,
                                  const float* __restrict__ ctx, ushort_t* __restrict__ w2,
                                  float* __restrict__ bias) {
  int idx = blockIdx.x * blockDim.x + threadIdx.x;  // b*16384 + c*128 + hd
  if (idx >= Bb * HIDc * HIDc) return;
  int b = idx >> 14;
  int c = (idx >> 7) & 127;
  int hd = idx & 127;
  int h = hd >> 5, d = hd & 31;
  const float* ow = out_w + (long long)c * HIDc + h * DHEAD;
  const float* cx = ctx + ((long long)(b * 4 + h) * 32 + d) * 32;
  float s = 0.f;
  for (int e = 0; e < 32; ++e) s += ow[e] * cx[e];
  w2[idx] = f2bf(s);
  if (hd == 0) bias[b * HIDc + c] = out_b[c];
}

// ---------------- final: GroupNorm(po) * w + b + residual x ----------------
__global__ void final_kernel(const float* __restrict__ po, const float* __restrict__ x,
                             const float* __restrict__ stats2, const float* __restrict__ onw,
                             const float* __restrict__ onb, float* __restrict__ out) {
  long long id = (long long)blockIdx.x * blockDim.x + threadIdx.x;  // b*2^24 + c*2^17 + n
  int b = (int)(id >> 24);
  int c = (int)((id >> 17) & 127);
  int g = c >> 2;
  float mean = stats2[(b * 32 + g) * 2];
  float rstd = stats2[(b * 32 + g) * 2 + 1];
  out[id] = (po[id] - mean) * rstd * onw[c] + onb[c] + x[id];
}

extern "C" void kernel_launch(void* const* d_in, const int* in_sizes, int n_in,
                              void* d_out, int out_size, void* d_ws, size_t ws_size,
                              hipStream_t stream) {
  (void)in_sizes; (void)n_in; (void)out_size; (void)ws_size;
  const float* x          = (const float*)d_in[0];
  const float* norm_w     = (const float*)d_in[1];
  const float* norm_b     = (const float*)d_in[2];
  const float* qkv_w      = (const float*)d_in[3];
  const float* out_w      = (const float*)d_in[4];
  const float* out_b      = (const float*)d_in[5];
  const float* out_norm_w = (const float*)d_in[6];
  const float* out_norm_b = (const float*)d_in[7];

  char* ws = (char*)d_ws;
  size_t off = 0;
  auto alloc = [&](size_t bytes) -> void* {
    void* p = ws + off;
    off = (off + bytes + 255) & ~(size_t)255;
    return p;
  };
  ushort_t* qkvbf = (ushort_t*)alloc((size_t)Bb * MQKV * Nn * 2);   // 201 MB (q|k|v bf16)
  float*    po    = (float*)   alloc((size_t)Bb * HIDc * Nn * 4);   // 134 MB
  float*    st1   = (float*)   alloc((size_t)Bb * 32 * 2 * 4);
  float*    qst   = (float*)   alloc((size_t)Bb * HIDc * 2 * 4);
  float*    part  = (float*)   alloc((size_t)8 * NSLICES * 1024 * 4);
  float*    ctx   = (float*)   alloc((size_t)8 * 1024 * 4);
  ushort_t* wq    = (ushort_t*)alloc((size_t)Bb * MQKV * Cc * 2);
  float*    bq    = (float*)   alloc((size_t)Bb * MQKV * 4);
  ushort_t* w2    = (ushort_t*)alloc((size_t)Bb * HIDc * HIDc * 2);
  float*    b2    = (float*)   alloc((size_t)Bb * HIDc * 4);
  float*    st2   = (float*)   alloc((size_t)Bb * 32 * 2 * 4);

  // 1. GroupNorm stats on x
  gn_stats_kernel<<<Bb * GROUPSc, 256, 0, stream>>>(x, st1);
  // 2. fold norm into qkv weights
  prep_qkv_w_kernel<<<(Bb * MQKV + 255) / 256, 256, 0, stream>>>(qkv_w, norm_w, norm_b, st1, wq, bq);
  // 3. qkv GEMM: [384 x 128] x [128 x N] -> bf16
  dim3 g1(Nn / NB, Bb);
  wmma_gemm_kernel<<<g1, 256, 0, stream>>>(wq, bq, (const void*)x, 1, (long long)Cc * Nn,
                                           (void*)qkvbf, 0, (long long)MQKV * Nn, MQKV);
  // 4. k softmax over d (in place)
  ksoftmax_kernel<<<(Bb * HEADSc * Nn) / 256, 256, 0, stream>>>(qkvbf);
  // 5/6. q softmax over N (stats + in-place normalize)
  qstats_kernel<<<Bb * HIDc, 256, 0, stream>>>(qkvbf, qst);
  qnorm_kernel<<<(Bb * HIDc * Nn) / 256, 256, 0, stream>>>(qkvbf, qst);
  // 7/8. ctx = k' v^T  (slice partials + deterministic reduce)
  dim3 g2(NSLICES, Bb * HEADSc);
  ctx_partial_kernel<<<g2, 256, 0, stream>>>(qkvbf, part);
  ctx_reduce_kernel<<<(8 * 1024) / 256, 256, 0, stream>>>(part, ctx);
  // 9. fold ctx into out_w
  prep_out_w_kernel<<<(Bb * HIDc * HIDc + 255) / 256, 256, 0, stream>>>(out_w, out_b, ctx, w2, b2);
  // 10. output GEMM: [128 x 128] x [128 x N] -> f32
  wmma_gemm_kernel<<<g1, 256, 0, stream>>>(w2, b2, (const void*)qkvbf, 0, (long long)MQKV * Nn,
                                           (void*)po, 1, (long long)HIDc * Nn, HIDc);
  // 11. GroupNorm stats on projection output
  gn_stats_kernel<<<Bb * GROUPSc, 256, 0, stream>>>(po, st2);
  // 12. normalize + residual
  final_kernel<<<(Bb * HIDc * Nn) / 256, 256, 0, stream>>>(po, x, st2, out_norm_w, out_norm_b,
                                                           (float*)d_out);
}
